// FSSLoss_53498112639530
// MI455X (gfx1250) — matrix operand order repaired and checked by reading, not compiled, not run on hardware
//
#include <hip/hip_runtime.h>
#include <hip/hip_bf16.h>

typedef __attribute__((ext_vector_type(16))) _Float16 v16h;
typedef __attribute__((ext_vector_type(8)))  float    v8f;

#define B_    16
#define S_    12
#define H_    480
#define W_    480
#define OUTHW 473     // H - 8 + 1
#define TILES 30      // ceil(473/16)

// 0.5*tanh(a)+0.5 == sigmoid(2a) == 1/(1+2^c), c = -2a*log2(e)
__device__ __forceinline__ float sigmoid2exp(float c) {
    float e = __builtin_amdgcn_exp2f(c);        // v_exp_f32 (inf -> rcp gives 0)
    return __builtin_amdgcn_rcpf(1.0f + e);     // v_rcp_f32
}

// ---------------- Kernel 0: zero the per-batch accumulators --------------
__global__ void fss_zero(float* __restrict__ acc) {
    if (threadIdx.x < 3 * B_) acc[threadIdx.x] = 0.0f;
}

// ---- Kernel 1: fused pixel-scale + tanh-threshold + channel(S)-sum ------
// Streams the full 354 MB of input once; writes two 14.7 MB f32 fields.
__global__ void fss_chansum(const float* __restrict__ outp,
                            const float* __restrict__ tgtp,
                            float* __restrict__ fo,
                            float* __restrict__ ft) {
    const int HW = H_ * W_;
    int p = blockIdx.x * blockDim.x + threadIdx.x;
    if (p >= B_ * HW) return;
    int b   = p / HW;
    int rem = p - b * HW;
    size_t base = (size_t)b * S_ * HW + rem;
    // c = -40*log2(e) * (255*x - 74)
    const float K = -57.70780163555854f;
    float so = 0.0f, st = 0.0f;
#pragma unroll
    for (int s = 0; s < S_; ++s) {
        float xo = outp[base + (size_t)s * HW];
        float xt = tgtp[base + (size_t)s * HW];
        so += sigmoid2exp(K * __builtin_fmaf(xo, 255.0f, -74.0f));
        st += sigmoid2exp(K * __builtin_fmaf(xt, 255.0f, -74.0f));
    }
    fo[p] = so;
    ft[p] = st;
}

// ---- Kernel 2: 8x8 box filter via WMMA band-matmul + FSS reductions -----
// One wave per 16x16 output tile. Horizontal 8-sums from LDS, vertical
// 8-sum as D = A_band(16x32,f16) x H(32x16,f16) with f32 accumulate.
// Boundary handling is branchless: invalid columns are zeroed in B (so they
// contribute exactly 0 to all sums); invalid rows get a 0/1 weight multiply.
__global__ void __launch_bounds__(32)
fss_box_wmma(const float* __restrict__ fo,
             const float* __restrict__ ft,
             float* __restrict__ acc) {
    __shared__ float so_s[24][25];   // +1 pad vs 64-bank LDS
    __shared__ float st_s[24][25];

    const int lane = threadIdx.x;            // wave32
    const int t  = blockIdx.x;
    const int tx = t % TILES;
    const int ty = (t / TILES) % TILES;
    const int b  = t / (TILES * TILES);
    const int i0 = ty * 16, j0 = tx * 16;

    const float* fob = fo + (size_t)b * H_ * W_;
    const float* ftb = ft + (size_t)b * H_ * W_;

    // Stage 24 rows x 24 cols (rows i0..i0+23, cols j0..j0+23), zero-padded.
    if (lane < 24) {
        int col = j0 + lane;
        bool cok = (col < W_);
#pragma unroll 4
        for (int r = 0; r < 24; ++r) {
            int row = i0 + r;
            bool ok = cok && (row < H_);
            size_t off = (size_t)row * W_ + col;
            so_s[r][lane] = ok ? fob[off] : 0.0f;
            st_s[r][lane] = ok ? ftb[off] : 0.0f;
        }
    }
    __syncthreads();

    const int m  = lane & 15;
    const int hi = lane >> 4;   // 0: lanes 0-15, 1: lanes 16-31

    // A = band matrix: A[m][k] = 1 iff m <= k < m+8 (documented 16-bit A layout:
    // element e -> K = e + (e<8 ? 0 : 8) + (hi ? 8 : 0), M = lane&15).
    v16h a;
#pragma unroll
    for (int e = 0; e < 16; ++e) {
        int k = (e < 8 ? e : e + 8) + (hi ? 8 : 0);
        a[e] = (k >= m && k < m + 8) ? (_Float16)1.0f : (_Float16)0.0f;
    }

    // B = horizontal 8-sums H[k][n] (B layout: column n = lane&15 split across
    // lane/lane+16; element e -> K = e + (e<8 ? 0 : 8) + (hi ? 8 : 0)). Only k<=22 is live.
    // Columns past the 473 output edge are zeroed here (branchless epilogue).
    const float colw = ((j0 + m) < OUTHW) ? 1.0f : 0.0f;
    v16h bo, bt;
#pragma unroll
    for (int e = 0; e < 16; ++e) {
        int k = (e < 8 ? e : e + 8) + (hi ? 8 : 0);
        float ho = 0.0f, ht = 0.0f;
        if (k < 23) {
#pragma unroll
            for (int q = 0; q < 8; ++q) {
                ho += so_s[k][m + q];
                ht += st_s[k][m + q];
            }
        }
        bo[e] = (_Float16)(ho * colw);
        bt[e] = (_Float16)(ht * colw);
    }

    v8f c0 = {};
    v8f dco = __builtin_amdgcn_wmma_f32_16x16x32_f16(false, a, false, bo,
                                                     (short)0, c0, false, false);
    v8f dct = __builtin_amdgcn_wmma_f32_16x16x32_f16(false, a, false, bt,
                                                     (short)0, c0, false, false);

    // C/D layout: lane holds column n = lane&15; VGPR v = row v + (hi?8:0).
    // Row mask as a 0/1 multiplier -> no exec-mask divergence.
    float num = 0.0f, s2o = 0.0f, s2t = 0.0f;
#pragma unroll
    for (int v = 0; v < 8; ++v) {
        int orow = i0 + v + (hi ? 8 : 0);
        float w  = (orow < OUTHW) ? 1.0f : 0.0f;
        float oo = dco[v] * (0.015625f * w);   // /64, masked
        float tt = dct[v] * (0.015625f * w);
        float d  = oo - tt;
        num = __builtin_fmaf(d,  d,  num);
        s2o = __builtin_fmaf(oo, oo, s2o);
        s2t = __builtin_fmaf(tt, tt, s2t);
    }

    // wave32 reduction
#pragma unroll
    for (int off = 16; off >= 1; off >>= 1) {
        num += __shfl_xor(num, off, 32);
        s2o += __shfl_xor(s2o, off, 32);
        s2t += __shfl_xor(s2t, off, 32);
    }
    if (lane == 0) {
        atomicAdd(&acc[b * 3 + 0], num);
        atomicAdd(&acc[b * 3 + 1], s2o);
        atomicAdd(&acc[b * 3 + 2], s2t);
    }
}

// ---------------- Kernel 3: final scalar = mean_b(num/den) ---------------
__global__ void fss_final(const float* __restrict__ acc, float* __restrict__ out) {
    int lane = threadIdx.x;
    float q = 0.0f;
    if (lane < B_) {
        float num = acc[lane * 3 + 0];
        float den = acc[lane * 3 + 1] + acc[lane * 3 + 2];
        q = num / den;
    }
#pragma unroll
    for (int off = 8; off >= 1; off >>= 1) q += __shfl_xor(q, off, 32);
    if (lane == 0) out[0] = q * (1.0f / (float)B_);
}

extern "C" void kernel_launch(void* const* d_in, const int* in_sizes, int n_in,
                              void* d_out, int out_size, void* d_ws, size_t ws_size,
                              hipStream_t stream) {
    (void)in_sizes; (void)n_in; (void)out_size; (void)ws_size;
    const float* outp = (const float*)d_in[0];
    const float* tgtp = (const float*)d_in[1];

    // Workspace layout (floats): fo[B*H*W] | ft[B*H*W] | acc[B*3]  (~29.5 MB)
    float* ws  = (float*)d_ws;
    float* fo  = ws;
    float* ft  = ws + (size_t)B_ * H_ * W_;
    float* acc = ft + (size_t)B_ * H_ * W_;

    fss_zero<<<1, 64, 0, stream>>>(acc);

    const int total = B_ * H_ * W_;
    fss_chansum<<<(total + 255) / 256, 256, 0, stream>>>(outp, tgtp, fo, ft);

    fss_box_wmma<<<B_ * TILES * TILES, 32, 0, stream>>>(fo, ft, acc);

    fss_final<<<1, 32, 0, stream>>>(acc, (float*)d_out);
}
